// TextualKnowledgeInjector_71270687309839
// MI455X (gfx1250) — matrix operand order repaired and checked by reading, not compiled, not run on hardware
//
#include <hip/hip_runtime.h>

// ---------------------------------------------------------------------------
// Problem constants (reference: B=64, T=256, F=40, D=768, H=256)
// ---------------------------------------------------------------------------
#define BT   16384      // B*T rows
#define FF   40         // features
#define KK   1600       // F*F flattened pair index
#define DD   768        // embedding dim
#define HH   256        // adapter out dim

typedef __attribute__((ext_vector_type(16))) _Float16 v16h;
typedef __attribute__((ext_vector_type(8)))  _Float16 v8h;
typedef __attribute__((ext_vector_type(8)))  float    v8f;

// Workspace layout (bytes, all 16B-aligned). Total ~28.2 MB.
#define OFF_ET   0u          // Et  [DD][KK] fp16 : 2,457,600 B
#define OFF_W16  2457600u    // W16 [HH][DD] fp16 :   393,216 B
#define OFF_RM   2850816u    // rowmask [BT] u64  :   131,072 B
#define OFF_AM   2981888u    // availmask[FF] u64 :       512 B (padded)
#define OFF_CTX  2982400u    // ctx16 [BT][DD] fp16: 25,165,824 B

// ---------------------------------------------------------------------------
// Async global->LDS copy (CDNA5 ASYNCcnt path) with safe fallback.
// The async load is emitted with inline asm: VDST = per-lane LDS byte address
// (low 32 bits of the generic pointer into __shared__), VADDR = 64-bit global
// address. Completion is tracked with ASYNCcnt -> s_wait_asynccnt.
// ---------------------------------------------------------------------------
#if defined(__gfx1250__) && defined(__has_builtin)
#if __has_builtin(__builtin_amdgcn_s_wait_asynccnt)
#define ATHENA_ASYNC_LDS 1
#endif
#endif

__device__ __forceinline__ void copy16_to_lds(_Float16* lds_dst, const _Float16* gsrc) {
#ifdef ATHENA_ASYNC_LDS
  const unsigned lds_off = (unsigned)(unsigned long long)lds_dst;  // LDS byte addr
  asm volatile("global_load_async_to_lds_b128 %0, %1, off"
               :
               : "v"(lds_off), "v"(gsrc)
               : "memory");
#else
  *(v8h*)lds_dst = *(const v8h*)gsrc;
#endif
}

__device__ __forceinline__ void async_copy_fence() {
#ifdef ATHENA_ASYNC_LDS
  __builtin_amdgcn_s_wait_asynccnt(0);
#endif
}

// ---------------------------------------------------------------------------
// Prep: pack surviving_mask rows and avail(i,j & i<j) rows into u64 bitmasks
// ---------------------------------------------------------------------------
__global__ __launch_bounds__(256)
void prep_masks(const int* __restrict__ surv, const int* __restrict__ pavail,
                unsigned long long* __restrict__ rowmask,
                unsigned long long* __restrict__ availmask) {
  const int tid = blockIdx.x * blockDim.x + threadIdx.x;
  if (tid < BT) {
    unsigned long long m = 0ull;
    #pragma unroll 8
    for (int f = 0; f < FF; ++f)
      m |= (unsigned long long)(surv[tid * FF + f] & 1) << f;
    rowmask[tid] = m;
  }
  if (tid < FF) {
    unsigned long long a = 0ull;
    for (int j = tid + 1; j < FF; ++j)
      a |= (unsigned long long)(pavail[tid * FF + j] & 1) << j;
    availmask[tid] = a;
  }
}

// ---------------------------------------------------------------------------
// Prep: Et[d][k] = fp16( avail(i<j) ? pair_emb[i][j][d] : 0 ), k = i*40+j.
// K-major rows so each lane's WMMA B-fragment is two contiguous 8-half k runs.
// ---------------------------------------------------------------------------
__global__ __launch_bounds__(256)
void prep_et(const float* __restrict__ pemb, const int* __restrict__ pavail,
             _Float16* __restrict__ Et) {
  const int tid = blockIdx.x * blockDim.x + threadIdx.x;   // KK*DD threads
  const int k = tid / DD;
  const int d = tid - k * DD;
  const int i = k / FF;
  const int j = k - i * FF;
  float v = 0.0f;
  if (i < j && (pavail[k] & 1)) v = pemb[(size_t)k * DD + d];
  Et[(size_t)d * KK + k] = (_Float16)v;
}

__global__ __launch_bounds__(256)
void prep_w16(const float* __restrict__ aw, _Float16* __restrict__ W16) {
  const int tid = blockIdx.x * blockDim.x + threadIdx.x;   // HH*DD threads
  W16[tid] = (_Float16)aw[tid];
}

// ---------------------------------------------------------------------------
// Generate 8 consecutive P entries P[row, k] = m_i & m_j for k = kstart..+7,
// as fp16 (exact 0.0/1.0). Incremental (i,j) avoids per-element div/mod.
// ---------------------------------------------------------------------------
__device__ __forceinline__ v8h gen_p8(unsigned kstart, unsigned long long m) {
  unsigned i = kstart / FF;
  unsigned j = kstart - i * FF;
  v8h r;
  #pragma unroll
  for (int t = 0; t < 8; ++t) {
    const unsigned bit = (unsigned)((m >> i) & (m >> j) & 1ull);
    r[t] = bit ? (_Float16)1.0f : (_Float16)0.0f;
    if (++j == FF) { j = 0u; ++i; }
  }
  return r;
}

#define CAT16(lo, hi) __builtin_shufflevector(lo, hi, 0,1,2,3,4,5,6,7,8,9,10,11,12,13,14,15)

// ---------------------------------------------------------------------------
// Stage 1: ctx16[row][d] = fp16( scale(row) * sum_k P[row,k] * Et[d][k] )
// Block tile: M=64 (4 mtiles) x N=192 (12 ntiles). 8 waves: wave = (msub, nhalf),
// each owning 16 rows x 96 cols (6 WMMAs per k-step). B staged in LDS per
// 32-wide K slice with coalesced/async copies. Grid = 256*4 = 1024 blocks.
// ---------------------------------------------------------------------------
__global__ __launch_bounds__(256)
void ctx_gemm(const _Float16* __restrict__ Et,
              const unsigned long long* __restrict__ rowmask,
              const unsigned long long* __restrict__ availmask,
              _Float16* __restrict__ ctx16) {
  __shared__ __align__(16) _Float16 Bs[192 * 32];   // [d_local][k_local], 12 KB

  const int lane  = threadIdx.x & 31;
  const int wid   = threadIdx.x >> 5;
  const int msub  = wid & 3;                 // which 16-row sub-tile
  const int nhalf = wid >> 2;                // which 96-col half of the block N
  const int hs    = lane >> 4;               // wave half (0|1)
  const int nl    = lane & 15;

  const int mblk  = blockIdx.x >> 2;         // 0..255
  const int nblk  = (blockIdx.x & 3) * 192;  // d base of block
  const int rbase = mblk * 64 + msub * 16;
  const int wnb   = nhalf * 96;              // wave col base inside block

  // Row mask for this lane's A row (lane L holds row L&15 per 16-bit A layout).
  const unsigned long long m64 = rowmask[rbase + nl];
  const unsigned c = (unsigned)hs * 8u;      // A K-chunk offset for upper lanes

  v8f acc[6] = {};
  for (unsigned kb = 0; kb < KK; kb += 32) {
    __syncthreads();                         // previous slice fully consumed
    // Stage Et[nblk..nblk+191][kb..kb+31] -> LDS (3 x 16B per thread, 64B/row)
    #pragma unroll
    for (int r = 0; r < 3; ++r) {
      const int chunk = (int)threadIdx.x + r * 256;    // 0..767
      const int row   = chunk >> 2;                    // 0..191 (a d index)
      const int kc    = (chunk & 3) * 8;
      const _Float16* src = Et + (size_t)(nblk + row) * KK + kb + kc;
      copy16_to_lds(&Bs[row * 32 + kc], src);
      __builtin_prefetch(src + 32, 0, 0);              // next K slice
    }
    async_copy_fence();
    __syncthreads();

    // A fragment: K local = {c..c+7, 16+c..16+c+7}, synthesized from bitmask
    const v8h alo = gen_p8(kb + c, m64);
    const v8h ahi = gen_p8(kb + 16u + c, m64);
    const v16h a = CAT16(alo, ahi);

    #pragma unroll
    for (int nt = 0; nt < 6; ++nt) {
      const int off = (wnb + nt * 16 + nl) * 32 + hs * 8;
      const v8h blo = *(const v8h*)&Bs[off];
      const v8h bhi = *(const v8h*)&Bs[off + 16];
      const v16h b = CAT16(blo, bhi);
      acc[nt] = __builtin_amdgcn_wmma_f32_16x16x32_f16(
          false, a, false, b, (short)0, acc[nt], false, false);
    }
  }

  // cnt[row] = sum_i m_i * popcount(m & avail_i);  scale = cnt>0 ? 1/cnt : 0
  unsigned icnt = 0;
  #pragma unroll 8
  for (int i = 0; i < FF; ++i)
    icnt += (unsigned)((m64 >> i) & 1ull) * (unsigned)__popcll(m64 & availmask[i]);
  const float scale = icnt ? (1.0f / (float)icnt) : 0.0f;

  // C layout: VGPR v, lane L -> M = v + 8*(L>=16), N = L&15
  #pragma unroll
  for (int v = 0; v < 8; ++v) {
    const float s = __shfl(scale, v + hs * 8, 32);
    const int row = rbase + v + hs * 8;
    #pragma unroll
    for (int nt = 0; nt < 6; ++nt) {
      const int col = nblk + wnb + nt * 16 + nl;
      ctx16[(size_t)row * DD + col] = (_Float16)(acc[nt][v] * s);
    }
  }
}

// ---------------------------------------------------------------------------
// Stage 2: out[row][h] = ctx[row][:] . W[h][:] + b[h]   (M=16384,N=256,K=768)
// Block tile: M=64 x N=128; wave = 16 rows x 64 cols. A and B staged in LDS.
// Grid = 256*2 = 512 blocks.
// ---------------------------------------------------------------------------
__global__ __launch_bounds__(256)
void adapter_gemm(const _Float16* __restrict__ ctx16,
                  const _Float16* __restrict__ W16,
                  const float* __restrict__ bias,
                  float* __restrict__ out) {
  __shared__ __align__(16) _Float16 As[64 * 32];    // [row_local][k_local], 4 KB
  __shared__ __align__(16) _Float16 Bs[128 * 32];   // [h_local][k_local],  8 KB

  const int lane  = threadIdx.x & 31;
  const int wid   = threadIdx.x >> 5;
  const int msub  = wid & 3;
  const int nhalf = wid >> 2;
  const int hs    = lane >> 4;
  const int nl    = lane & 15;

  const int mblk  = blockIdx.x >> 1;          // 0..255
  const int nb    = (blockIdx.x & 1) * 128;   // h base of block
  const int rbase = mblk * 64 + msub * 16;
  const int wnb   = nhalf * 64;               // wave col base inside block
  const unsigned c = (unsigned)hs * 8u;

  v8f acc[4] = {};
  for (unsigned kb = 0; kb < DD; kb += 32) {
    __syncthreads();
    // Stage A: ctx16[mblk*64 .. +63][kb..kb+31] (1 x 16B per thread)
    {
      const int row = (int)threadIdx.x >> 2;             // 0..63
      const int kc  = ((int)threadIdx.x & 3) * 8;
      const _Float16* src = ctx16 + (size_t)(mblk * 64 + row) * DD + kb + kc;
      copy16_to_lds(&As[row * 32 + kc], src);
      __builtin_prefetch(src + 32, 0, 0);
    }
    // Stage B: W16[nb .. nb+127][kb..kb+31] (2 x 16B per thread)
    #pragma unroll
    for (int r = 0; r < 2; ++r) {
      const int chunk = (int)threadIdx.x + r * 256;      // 0..511
      const int row   = chunk >> 2;                      // 0..127 (an h index)
      const int kc    = (chunk & 3) * 8;
      const _Float16* src = W16 + (size_t)(nb + row) * DD + kb + kc;
      copy16_to_lds(&Bs[row * 32 + kc], src);
      __builtin_prefetch(src + 32, 0, 0);
    }
    async_copy_fence();
    __syncthreads();

    const int aoff = (msub * 16 + nl) * 32 + (int)c;
    const v8h alo = *(const v8h*)&As[aoff];
    const v8h ahi = *(const v8h*)&As[aoff + 16];
    const v16h a = CAT16(alo, ahi);

    #pragma unroll
    for (int nt = 0; nt < 4; ++nt) {
      const int off = (wnb + nt * 16 + nl) * 32 + hs * 8;
      const v8h blo = *(const v8h*)&Bs[off];
      const v8h bhi = *(const v8h*)&Bs[off + 16];
      const v16h b = CAT16(blo, bhi);
      acc[nt] = __builtin_amdgcn_wmma_f32_16x16x32_f16(
          false, a, false, b, (short)0, acc[nt], false, false);
    }
  }

  float bv[4];
  #pragma unroll
  for (int nt = 0; nt < 4; ++nt) bv[nt] = bias[nb + wnb + nt * 16 + nl];

  #pragma unroll
  for (int v = 0; v < 8; ++v) {
    const int row = rbase + v + hs * 8;
    #pragma unroll
    for (int nt = 0; nt < 4; ++nt) {
      const int col = nb + wnb + nt * 16 + nl;
      out[(size_t)row * HH + col] = acc[nt][v] + bv[nt];
    }
  }
}

// ---------------------------------------------------------------------------
// Launch
// ---------------------------------------------------------------------------
extern "C" void kernel_launch(void* const* d_in, const int* in_sizes, int n_in,
                              void* d_out, int out_size, void* d_ws, size_t ws_size,
                              hipStream_t stream) {
  const int*   surv   = (const int*)d_in[0];    // surviving_mask [B,T,F] bool->int
  const float* pemb   = (const float*)d_in[1];  // pair_emb [F,F,D] f32
  const int*   pavail = (const int*)d_in[2];    // pair_avail [F,F] bool->int
  const float* aw     = (const float*)d_in[3];  // adapter_w [H,D] f32
  const float* ab     = (const float*)d_in[4];  // adapter_b [H] f32
  float* out = (float*)d_out;                   // [B,T,H] f32

  char* ws = (char*)d_ws;
  _Float16*           Et  = (_Float16*)(ws + OFF_ET);
  _Float16*           W16 = (_Float16*)(ws + OFF_W16);
  unsigned long long* rm  = (unsigned long long*)(ws + OFF_RM);
  unsigned long long* am  = (unsigned long long*)(ws + OFF_AM);
  _Float16*           ctx = (_Float16*)(ws + OFF_CTX);

  prep_masks<<<BT / 256, 256, 0, stream>>>(surv, pavail, rm, am);
  prep_et<<<(KK * DD) / 256, 256, 0, stream>>>(pemb, pavail, Et);
  prep_w16<<<(HH * DD) / 256, 256, 0, stream>>>(aw, W16);
  // M: 16384/64 = 256 blocks, N: 768/192 = 4 -> 1024 blocks
  ctx_gemm<<<1024, 256, 0, stream>>>(Et, rm, am, ctx);
  // M: 16384/64 = 256 blocks, N: 256/128 = 2 -> 512 blocks
  adapter_gemm<<<512, 256, 0, stream>>>(ctx, W16, ab, out);
}
